// FP8QDQConv2d_65575560675872
// MI455X (gfx1250) — 8-bit matrix operand order repaired and checked by reading, compiled, not run on hardware
//
#include <hip/hip_runtime.h>
#include <hip/hip_bf16.h>


typedef __attribute__((ext_vector_type(16))) int   v16i;
typedef __attribute__((ext_vector_type(8)))  float v8f;
typedef __attribute__((ext_vector_type(4)))  int   v4i;

#define N_BATCH 32
#define C_IN    128
#define C_OUT   256
#define HW      56
#define NPIX    (HW * HW)
#define PTILES32 (NPIX / 32)
#define QX_BYTES ((size_t)N_BATCH * NPIX * C_IN)
#define QW_BYTES ((size_t)9 * 16 * 2048)
#define QW_OFF   (QX_BYTES + 256)
#define PANEL_BYTES 32768
#define FP8_MAX 448.0f

#if __has_builtin(__builtin_amdgcn_cvt_pk_fp8_f32)
#define HAVE_CVT_FP8 1
#else
#define HAVE_CVT_FP8 0
#endif

#if __has_builtin(__builtin_amdgcn_global_load_async_to_lds_b128)
#define HAVE_ASYNC_LDS 1
#else
#define HAVE_ASYNC_LDS 0
#endif

#if __has_builtin(__builtin_amdgcn_s_wait_asynccnt)
#define WAIT_ASYNC0() __builtin_amdgcn_s_wait_asynccnt(0)
#else
#define WAIT_ASYNC0() asm volatile("s_wait_asynccnt 0x0" ::: "memory")
#endif

typedef __attribute__((address_space(1))) v4i glb_v4i;
typedef __attribute__((address_space(3))) v4i lds_v4i;

__device__ __forceinline__ unsigned int f32_to_e4m3(float fin) {
  unsigned int x    = __float_as_uint(fin);
  unsigned int sign = (x >> 24) & 0x80u;
  x &= 0x7FFFFFFFu;
  if (x < 0x3C800000u) {
    float f = __uint_as_float(x);
    unsigned int q = (unsigned int)__builtin_rintf(f * 512.0f);
    return sign | q;
  }
  unsigned int exp  = x >> 23;
  unsigned int mant = x & 0x7FFFFFu;
  unsigned int m3   = mant >> 20;
  unsigned int rem  = mant & 0xFFFFFu;
  unsigned int rnd  = (rem > 0x80000u) || ((rem == 0x80000u) && (m3 & 1u));
  unsigned int code = (((exp - 120u) << 3) | m3) + rnd;
  if (code > 0x7Eu) code = 0x7Eu;
  return sign | code;
}

__global__ __launch_bounds__(256) void quantize_x_nhwc(
    const float* __restrict__ x, unsigned char* __restrict__ qx) {
  unsigned int tid = blockIdx.x * 256u + threadIdx.x;
  unsigned int c4  = tid & 31u;
  unsigned int pix = tid >> 5;
  unsigned int n   = pix / NPIX;
  unsigned int hw  = pix - n * NPIX;
  const float* src = x + (size_t)n * C_IN * NPIX + (size_t)(c4 * 4u) * NPIX + hw;
  float f0 = src[0];
  float f1 = src[NPIX];
  float f2 = src[2 * NPIX];
  float f3 = src[3 * NPIX];
  f0 = fminf(fmaxf(f0 * 0.5f, -FP8_MAX), FP8_MAX);
  f1 = fminf(fmaxf(f1 * 0.5f, -FP8_MAX), FP8_MAX);
  f2 = fminf(fmaxf(f2 * 0.5f, -FP8_MAX), FP8_MAX);
  f3 = fminf(fmaxf(f3 * 0.5f, -FP8_MAX), FP8_MAX);
#if HAVE_CVT_FP8
  int pk = 0;
  pk = __builtin_amdgcn_cvt_pk_fp8_f32(f0, f1, pk, false);
  pk = __builtin_amdgcn_cvt_pk_fp8_f32(f2, f3, pk, true);
#else
  int pk = (int)(f32_to_e4m3(f0) | (f32_to_e4m3(f1) << 8) |
                 (f32_to_e4m3(f2) << 16) | (f32_to_e4m3(f3) << 24));
#endif
  *(int*)(qx + (size_t)pix * C_IN + c4 * 4u) = pk;
}

__global__ void zero_pad_record(unsigned char* __restrict__ qx) {
  ((int*)(qx + QX_BYTES))[threadIdx.x] = 0;
}

__global__ __launch_bounds__(256) void pack_weights(
    const float* __restrict__ w, unsigned char* __restrict__ qwp) {
  unsigned int tid    = blockIdx.x * 256u + threadIdx.x;
  unsigned int blk    = tid >> 11;
  unsigned int within = tid & 2047u;
  unsigned int lane   = within >> 6;
  unsigned int v      = (within >> 2) & 15u;
  unsigned int b      = within & 3u;
  unsigned int t      = blk >> 4;
  unsigned int otile  = blk & 15u;
  unsigned int K      = (v >> 1) * 16u + ((lane >= 16u) ? 8u : 0u) + (v & 1u) * 4u + b;
  unsigned int o      = otile * 16u + (lane & 15u);
  unsigned int kh     = t / 3u;
  unsigned int kw     = t - kh * 3u;
  float wv = w[(((size_t)o * C_IN + K) * 3u + kh) * 3u + kw];
  wv = fminf(fmaxf(wv, -FP8_MAX), FP8_MAX);
#if HAVE_CVT_FP8
  unsigned int byte = (unsigned int)__builtin_amdgcn_cvt_pk_fp8_f32(wv, 0.0f, 0, false) & 0xFFu;
#else
  unsigned int byte = f32_to_e4m3(wv);
#endif
  qwp[tid] = (unsigned char)byte;
}

__device__ __forceinline__ void stage_b_panel(const unsigned char* __restrict__ src_panel,
                                              unsigned char* __restrict__ dst_panel,
                                              unsigned tid) {
  const unsigned char* s = src_panel + tid * 16u;
  unsigned char*       d = dst_panel + tid * 16u;
#pragma unroll
  for (int k = 0; k < 8; ++k) {
#if HAVE_ASYNC_LDS
    __builtin_amdgcn_global_load_async_to_lds_b128(
        (glb_v4i*)(uintptr_t)(s + k * 4096),
        (lds_v4i*)(unsigned)(uintptr_t)(d + k * 4096),
        0, 0);
#else
    *(v4i*)(d + k * 4096) = *(const v4i*)(s + k * 4096);
#endif
  }
}

__device__ __forceinline__ v16i load_a_frag(const unsigned char* __restrict__ qx,
                                            int pidx, int dkh, int dkw,
                                            int oh, int ow, int rowsel) {
  const int ih = oh + dkh;
  const int iw = ow + dkw;
  const bool valid = ((unsigned)ih < (unsigned)HW) && ((unsigned)iw < (unsigned)HW);
  const unsigned real = (unsigned)(pidx + dkh * HW + dkw) * (unsigned)C_IN;
  const unsigned off  = (valid ? real : (unsigned)QX_BYTES) + (unsigned)(rowsel * 8);
  const unsigned char* rb = qx + off;
  v16i a;
#pragma unroll
  for (int j = 0; j < 8; ++j) {
    uint2 dd = *(const uint2*)(rb + j * 16);
    a[2 * j]     = (int)dd.x;
    a[2 * j + 1] = (int)dd.y;
  }
  return a;
}

__device__ __forceinline__ v16i load_b_frag_lds(const unsigned char* bb) {
  v16i bm;
#pragma unroll
  for (int q = 0; q < 4; ++q) {
    v4i u = *(const v4i*)(bb + q * 16);
    bm[4 * q]     = u.x;
    bm[4 * q + 1] = u.y;
    bm[4 * q + 2] = u.z;
    bm[4 * q + 3] = u.w;
  }
  return bm;
}

__global__ __launch_bounds__(256) void conv3x3_fp8_wmma(
    const unsigned char* __restrict__ qx,
    const unsigned char* __restrict__ qwp,
    const float* __restrict__ bias,
    float* __restrict__ out) {
  __shared__ unsigned char bpanel[2][PANEL_BYTES];

  const unsigned tid  = threadIdx.x;
  const int lane   = tid & 31;
  const int wave   = tid >> 5;
  const int n      = blockIdx.x / PTILES32;
  const int p0     = (blockIdx.x % PTILES32) * 32;

  const int m      = lane & 15;
  const int rowsel = lane >> 4;

  const int pA  = p0 + m;
  const int pB  = pA + 16;
  const int ohA = pA / HW, owA = pA - ohA * HW;
  const int ohB = pB / HW, owB = pB - ohB * HW;
  const int piA = n * NPIX + pA;
  const int piB = n * NPIX + pB;

  const int ct0 = wave * 2;
  const unsigned bofs = (unsigned)(ct0 * 2048 + lane * 64);

  __builtin_prefetch(qx + (unsigned)piA * C_IN, 0, 1);

  stage_b_panel(qwp, &bpanel[0][0], tid);
  WAIT_ASYNC0();
  __syncthreads();

  v8f acc00 = {};
  v8f acc01 = {};
  v8f acc10 = {};
  v8f acc11 = {};

#pragma unroll
  for (int t = 0; t < 9; ++t) {
    const int kh  = t / 3;
    const int dkh = kh - 1;
    const int dkw = (t - kh * 3) - 1;

    if (t < 8)
      stage_b_panel(qwp + (size_t)(t + 1) * PANEL_BYTES, &bpanel[(t + 1) & 1][0], tid);

    const v16i a0 = load_a_frag(qx, piA, dkh, dkw, ohA, owA, rowsel);
    const v16i a1 = load_a_frag(qx, piB, dkh, dkw, ohB, owB, rowsel);

    const unsigned char* bb = &bpanel[t & 1][bofs];
    const v16i b0 = load_b_frag_lds(bb);
    const v16i b1 = load_b_frag_lds(bb + 2048);

    acc00 = __builtin_amdgcn_wmma_f32_16x16x128_fp8_fp8(a0, b0, (short)0, acc00, false, false);
    acc01 = __builtin_amdgcn_wmma_f32_16x16x128_fp8_fp8(a0, b1, (short)0, acc01, false, false);
    acc10 = __builtin_amdgcn_wmma_f32_16x16x128_fp8_fp8(a1, b0, (short)0, acc10, false, false);
    acc11 = __builtin_amdgcn_wmma_f32_16x16x128_fp8_fp8(a1, b1, (short)0, acc11, false, false);

    if (t < 8) {
      WAIT_ASYNC0();
      __syncthreads();
    }
  }

  const int ncol = lane & 15;
  const int co0  = ct0 * 16 + ncol;
  const int co1  = co0 + 16;
  const float bv0 = bias[co0];
  const float bv1 = bias[co1];

#pragma unroll
  for (int d = 0; d < 8; ++d) {
    const int M = d + rowsel * 8;
    {
      const int pp  = p0 + M;
      const int ohh = pp / HW;
      const int oww = pp - ohh * HW;
      const size_t base = ((size_t)n * C_OUT * HW + ohh) * HW + oww;
      out[base + (size_t)co0 * NPIX] = acc00[d] * 4.0f + bv0;
      out[base + (size_t)co1 * NPIX] = acc01[d] * 4.0f + bv1;
    }
    {
      const int pp  = p0 + 16 + M;
      const int ohh = pp / HW;
      const int oww = pp - ohh * HW;
      const size_t base = ((size_t)n * C_OUT * HW + ohh) * HW + oww;
      out[base + (size_t)co0 * NPIX] = acc10[d] * 4.0f + bv0;
      out[base + (size_t)co1 * NPIX] = acc11[d] * 4.0f + bv1;
    }
  }
}

extern "C" void kernel_launch(void* const* d_in, const int* in_sizes, int n_in,
                              void* d_out, int out_size, void* d_ws, size_t ws_size,
                              hipStream_t stream) {
  const float* x    = (const float*)d_in[0];
  const float* w    = (const float*)d_in[1];
  const float* bias = (const float*)d_in[2];
  float* out        = (float*)d_out;

  unsigned char* qx  = (unsigned char*)d_ws;
  unsigned char* qwp = (unsigned char*)d_ws + QW_OFF;

  quantize_x_nhwc<<<dim3((N_BATCH * NPIX * 32) / 256), dim3(256), 0, stream>>>(x, qx);

  zero_pad_record<<<dim3(1), dim3(32), 0, stream>>>(qx);

  pack_weights<<<dim3((unsigned)(QW_BYTES / 256)), dim3(256), 0, stream>>>(w, qwp);

  conv3x3_fp8_wmma<<<dim3(N_BATCH * PTILES32), dim3(256), 0, stream>>>(qx, qwp, bias, out);
}
